// Encoder_78194174591612
// MI455X (gfx1250) — compile-verified, matrix operand order checked by reading, not run on hardware
//
#include <hip/hip_runtime.h>
#include <hip/hip_bf16.h>

typedef __bf16 bf16;
typedef __attribute__((ext_vector_type(16))) __bf16 v16bf;
typedef __attribute__((ext_vector_type(8)))  float  v8f;

#define BM 128
#define BN 128
#define BK 32
#define LDS_STRIDE 64   // bf16 elems per LDS row = 128 bytes -> 32B-aligned fragments

// ---------- constants for this problem ----------
#define CB   512   // batch
#define CL   128   // sequence length (bucket_size)
#define CD   512   // input dim
#define CH   512   // hidden
#define CA   128   // action hidden
#define CDEPTH 3

__device__ inline bf16 f2bf(float x) {
    union { float f; unsigned u; } v; v.f = x;
    unsigned r = (v.u + 0x7FFFu + ((v.u >> 16) & 1u)) >> 16;
    unsigned short s = (unsigned short)r;
    bf16 b; __builtin_memcpy(&b, &s, 2); return b;
}

// ---------------------------------------------------------------------------
// Generic bf16-WMMA GEMM: C(M x N) = A(f32, M x K, lda) * Bt(bf16, stored N x K)
// Double-buffered LDS pipeline; B tiles staged with async global->LDS copies
// (ASYNCcnt), A tiles converted f32->bf16 through registers.
// mode 0: out0[perm(r)*N + c] = acc + bias[c]        (perm = (r%L)*Bb + r/L if xpose)
// mode 1: fused recurrent step, N = nsplit + asz:
//         c <  nsplit: out0[r*nsplit+c]      = tanhf(acc + addW[r*nsplit+c])
//         c >= nsplit: outPa[r*asz+(c-ns)]   = max(acc + addA[r*asz+(c-ns)], 0)
// ---------------------------------------------------------------------------
__global__ __launch_bounds__(256)
void gemm_bf16_kernel(const float* __restrict__ A, int lda,
                      const bf16* __restrict__ Bt, int K, int N,
                      int mode,
                      const float* __restrict__ bias,
                      float* __restrict__ out0,
                      int xpose, int L, int Bb,
                      const float* __restrict__ addW,
                      const float* __restrict__ addA,
                      float* __restrict__ outPa,
                      int nsplit, int asz)
{
    __shared__ __attribute__((aligned(32))) bf16 As[2][BM][LDS_STRIDE];
    __shared__ __attribute__((aligned(32))) bf16 Bs[2][BN][LDS_STRIDE];

    const int tid  = threadIdx.x;
    const int lane = tid & 31;
    const int wave = tid >> 5;
    const int wm   = wave >> 2;      // 0..1
    const int wn   = wave & 3;       // 0..3

    const int mBase = blockIdx.y * BM;
    const int nBase = blockIdx.x * BN;

    const int lr = tid >> 1;          // 0..127
    const int lc = (tid & 1) * 16;    // 0 or 16

    const float* aPtr = A  + (size_t)(mBase + lr) * lda + lc;
    const bf16*  bPtr = Bt + (size_t)(nBase + lr) * K   + lc;

    v8f acc[4][2] = {};
    float4 areg[4];

    // stage 0
    {
        const float* s = aPtr;
        areg[0] = *(const float4*)(s + 0);
        areg[1] = *(const float4*)(s + 4);
        areg[2] = *(const float4*)(s + 8);
        areg[3] = *(const float4*)(s + 12);
        unsigned ldsa = (unsigned)(size_t)&Bs[0][lr][lc];
        const bf16* g = bPtr;
        asm volatile("global_load_async_to_lds_b128 %0, %1, off"
                     :: "v"(ldsa), "v"(g) : "memory");
        asm volatile("global_load_async_to_lds_b128 %0, %1, off offset:16"
                     :: "v"(ldsa), "v"(g) : "memory");
#pragma unroll
        for (int j = 0; j < 4; ++j) {
            As[0][lr][lc + j * 4 + 0] = f2bf(areg[j].x);
            As[0][lr][lc + j * 4 + 1] = f2bf(areg[j].y);
            As[0][lr][lc + j * 4 + 2] = f2bf(areg[j].z);
            As[0][lr][lc + j * 4 + 3] = f2bf(areg[j].w);
        }
        asm volatile("s_wait_asynccnt 0x0" ::: "memory");
    }
    __syncthreads();

    for (int k0 = 0; k0 < K; k0 += BK) {
        const int  cur  = (k0 / BK) & 1;
        const int  nxt  = cur ^ 1;
        const bool more = (k0 + BK) < K;

        if (more) {
            // issue next-tile staging into the other buffer while we compute
            const float* s = aPtr + k0 + BK;
            areg[0] = *(const float4*)(s + 0);
            areg[1] = *(const float4*)(s + 4);
            areg[2] = *(const float4*)(s + 8);
            areg[3] = *(const float4*)(s + 12);
            unsigned ldsa = (unsigned)(size_t)&Bs[nxt][lr][lc];
            const bf16* g = bPtr + k0 + BK;
            asm volatile("global_load_async_to_lds_b128 %0, %1, off"
                         :: "v"(ldsa), "v"(g) : "memory");
            asm volatile("global_load_async_to_lds_b128 %0, %1, off offset:16"
                         :: "v"(ldsa), "v"(g) : "memory");
            if (k0 + 2 * BK < K) __builtin_prefetch(aPtr + k0 + 2 * BK, 0, 1);
        }

        v16bf afrag[4];
#pragma unroll
        for (int mi = 0; mi < 4; ++mi)
            afrag[mi] = *(const v16bf*)&As[cur][wm * 64 + mi * 16 + (lane & 15)][(lane >> 4) * 16];
#pragma unroll
        for (int ni = 0; ni < 2; ++ni) {
            v16bf bfrag = *(const v16bf*)&Bs[cur][wn * 32 + ni * 16 + (lane & 15)][(lane >> 4) * 16];
#pragma unroll
            for (int mi = 0; mi < 4; ++mi)
                acc[mi][ni] = __builtin_amdgcn_wmma_f32_16x16x32_bf16(
                    false, afrag[mi], false, bfrag, (short)0, acc[mi][ni], false, false);
        }

        if (more) {
#pragma unroll
            for (int j = 0; j < 4; ++j) {
                As[nxt][lr][lc + j * 4 + 0] = f2bf(areg[j].x);
                As[nxt][lr][lc + j * 4 + 1] = f2bf(areg[j].y);
                As[nxt][lr][lc + j * 4 + 2] = f2bf(areg[j].z);
                As[nxt][lr][lc + j * 4 + 3] = f2bf(areg[j].w);
            }
            asm volatile("s_wait_asynccnt 0x0" ::: "memory");
        }
        __syncthreads();
    }

    // --- epilogue ---
#pragma unroll
    for (int mi = 0; mi < 4; ++mi) {
#pragma unroll
        for (int ni = 0; ni < 2; ++ni) {
            const int rowBase = mBase + wm * 64 + mi * 16;
            const int colBase = nBase + wn * 32 + ni * 16;
#pragma unroll
            for (int v = 0; v < 8; ++v) {
                int r = rowBase + (lane >> 4) * 8 + v;
                int c = colBase + (lane & 15);
                float val = acc[mi][ni][v];
                if (mode == 0) {
                    val += bias[c];
                    size_t orow = xpose ? (size_t)(r % L) * Bb + (r / L) : (size_t)r;
                    out0[orow * (size_t)N + c] = val;
                } else {
                    if (c < nsplit) {
                        size_t idx = (size_t)r * nsplit + c;
                        out0[idx] = tanhf(val + addW[idx]);
                    } else {
                        size_t idx = (size_t)r * asz + (c - nsplit);
                        outPa[idx] = fmaxf(val + addA[idx], 0.0f);
                    }
                }
            }
        }
    }
}

// f32 (K x N) -> bf16 transposed (N x K), with row offset into combined buffers
__global__ void cvt_tr_kernel(const float* __restrict__ in, bf16* __restrict__ out,
                              int K, int N, int rowOff, int outLD)
{
    int i = blockIdx.x * blockDim.x + threadIdx.x;
    if (i >= K * N) return;
    int n = i % N, k = i / N;
    out[(size_t)(n + rowOff) * outLD + k] = f2bf(in[(size_t)k * N + n]);
}

__global__ void init_dm_eos_kernel(const int* __restrict__ mask,
                                   int* __restrict__ dmInit, int* __restrict__ eosBuf,
                                   int L, int Bb)
{
    int i = blockIdx.x * blockDim.x + threadIdx.x;
    if (i >= L * Bb) return;
    int t = i / Bb, r = i % Bb;
    int dmt = mask[(size_t)r * L + t];
    int dmn = (t + 1 < L) ? mask[(size_t)r * L + t + 1] : 0;
    dmInit[i] = dmt;
    eosBuf[i] = dmt * (1 - dmn);
}

__global__ void init_state_kernel(float* __restrict__ hCur, int* __restrict__ actCur,
                                  int* __restrict__ dmCur, int BH, int Bb)
{
    int i = blockIdx.x * blockDim.x + threadIdx.x;
    if (i < BH) hCur[i] = 0.0f;
    if (i < Bb) { actCur[i] = 0; dmCur[i] = 0; }
}

// one wave (32 lanes) per batch row: policy head + action logic + gated h update
__global__ __launch_bounds__(256)
void act_gate_kernel(const float* __restrict__ Pa, const float* __restrict__ Hn,
                     const float* __restrict__ xsIn,
                     const float* __restrict__ Wa2, const float* __restrict__ ba2,
                     const int* __restrict__ aprevActs, const int* __restrict__ dmsPrev,
                     const int* __restrict__ eosBuf,
                     int* __restrict__ actCur, int* __restrict__ dmCur,
                     int* __restrict__ ActsOut, int* __restrict__ NdmsOut,
                     float* __restrict__ hCur, float* __restrict__ HsOut,
                     int t, int L, int Bb, int H, int A, int isLast)
{
    int wave = threadIdx.x >> 5;
    int lane = threadIdx.x & 31;
    int r = blockIdx.x * 8 + wave;
    if (r >= Bb) return;

    float p0 = 0.0f, p1 = 0.0f;
    for (int j = lane; j < A; j += 32) {
        float pv = Pa[(size_t)r * A + j];
        p0 += pv * Wa2[j * 2 + 0];
        p1 += pv * Wa2[j * 2 + 1];
    }
#pragma unroll
    for (int off = 16; off >= 1; off >>= 1) {
        p0 += __shfl_xor(p0, off, 32);
        p1 += __shfl_xor(p1, off, 32);
    }
    float pol0 = expf(p0 + ba2[0]);
    float pol1 = expf(p1 + ba2[1]);
    int act = (pol0 >= pol1) ? 1 : 0;

    int a_prev = 0;
    if (aprevActs != nullptr && (t + 1) < L) a_prev = aprevActs[(size_t)(t + 1) * Bb + r];
    int dm_prev = dmsPrev[(size_t)t * Bb + r];
    int eos_t   = eosBuf[(size_t)t * Bb + r];
    int dm_tm1  = dmCur[r];
    int a_tm1   = actCur[r];

    if (a_prev > 0) act = 1;
    if (isLast)     act = 1;
    if (eos_t > 0)  act = 0;

    int both   = (1 - a_prev) * dm_prev * act * dm_tm1;
    int h_only = dm_tm1 * act * (a_prev + (1 - a_prev) * (1 - dm_prev));
    int x_only = dm_prev * (1 - a_prev) * (1 - act + act * (1 - dm_tm1));
    int new_dm = both + x_only + h_only;
    int act_out = (dm_prev > 0) ? act : a_tm1;
    new_dm = dm_prev * new_dm;

    const float* xrow = xsIn + ((size_t)t * Bb + r) * H;
    for (int c = lane; c < H; c += 32) {
        float hnew = Hn[(size_t)r * H + c];
        float hold = hCur[(size_t)r * H + c];
        float h = 0.0f;
        if (both > 0)   h = hnew;
        if (h_only > 0) h = hold;
        if (x_only > 0) h = xrow[c];
        if (!(dm_prev > 0)) h = hold;
        hCur[(size_t)r * H + c] = h;
        HsOut[((size_t)t * Bb + r) * H + c] = h;
    }
    if (lane == 0) {
        actCur[r] = act_out;
        dmCur[r]  = new_dm;
        ActsOut[(size_t)t * Bb + r] = act_out;
        NdmsOut[(size_t)t * Bb + r] = new_dm;
    }
}

__global__ void copy_out_kernel(const float* __restrict__ src, float* __restrict__ dst, int n)
{
    int i = blockIdx.x * blockDim.x + threadIdx.x;
    if (i < n) dst[i] = src[i];
}

// ---------------------------------------------------------------------------
extern "C" void kernel_launch(void* const* d_in, const int* in_sizes, int n_in,
                              void* d_out, int out_size, void* d_ws, size_t ws_size,
                              hipStream_t stream)
{
    (void)in_sizes; (void)n_in; (void)out_size; (void)ws_size;

    const float* x     = (const float*)d_in[0];
    const int*   mask  = (const int*)  d_in[2];
    const float* W_emb = (const float*)d_in[3];
    const float* b_emb = (const float*)d_in[4];
    const float* W     = (const float*)d_in[5];
    const float* U     = (const float*)d_in[6];
    const float* b     = (const float*)d_in[7];
    const float* Wa1   = (const float*)d_in[8];
    const float* Ua1   = (const float*)d_in[9];
    const float* ba1   = (const float*)d_in[10];
    const float* Wa2   = (const float*)d_in[11];
    const float* ba2   = (const float*)d_in[12];
    float* out = (float*)d_out;

    // ---- workspace carve-up ----
    char* ws = (char*)d_ws;
    size_t off = 0;
    auto alloc = [&](size_t bytes) { size_t o = off; off += (bytes + 255) & ~(size_t)255; return o; };

    const size_t LBH = (size_t)CL * CB * CH;
    const size_t LBA = (size_t)CL * CB * CA;

    float* XsA   = (float*)(ws + alloc(LBH * 4));
    float* XsB   = (float*)(ws + alloc(LBH * 4));
    float* Xw    = (float*)(ws + alloc(LBH * 4));
    float* Xa    = (float*)(ws + alloc(LBA * 4));
    float* Hn    = (float*)(ws + alloc((size_t)CB * CH * 4));
    float* Pa    = (float*)(ws + alloc((size_t)CB * CA * 4));
    float* hCur  = (float*)(ws + alloc((size_t)CB * CH * 4));
    int*   Acts0 = (int*)  (ws + alloc((size_t)CL * CB * 4));
    int*   Acts1 = (int*)  (ws + alloc((size_t)CL * CB * 4));
    int*   Ndms0 = (int*)  (ws + alloc((size_t)CL * CB * 4));
    int*   Ndms1 = (int*)  (ws + alloc((size_t)CL * CB * 4));
    int*   dmIni = (int*)  (ws + alloc((size_t)CL * CB * 4));
    int*   eosB  = (int*)  (ws + alloc((size_t)CL * CB * 4));
    int*   actCur= (int*)  (ws + alloc((size_t)CB * 4));
    int*   dmCur = (int*)  (ws + alloc((size_t)CB * 4));
    bf16*  WembT = (bf16*) (ws + alloc((size_t)CH * CD * 2));          // H x D
    bf16*  WT    = (bf16*) (ws + alloc((size_t)CH * CH * 2));          // H x H
    bf16*  UUa1T = (bf16*) (ws + alloc((size_t)(CH + CA) * CH * 2));   // 640 x 512
    bf16*  Wa1T  = (bf16*) (ws + alloc((size_t)CA * CH * 2));          // A x H

    // ---- weight prep: f32 -> bf16, transposed to N x K ----
    {
        int thr = 256;
        cvt_tr_kernel<<<(CD * CH + thr - 1) / thr, thr, 0, stream>>>(W_emb, WembT, CD, CH, 0, CD);
        cvt_tr_kernel<<<(CH * CH + thr - 1) / thr, thr, 0, stream>>>(W,     WT,    CH, CH, 0, CH);
        cvt_tr_kernel<<<(CH * CH + thr - 1) / thr, thr, 0, stream>>>(U,     UUa1T, CH, CH, 0, CH);
        cvt_tr_kernel<<<(CH * CA + thr - 1) / thr, thr, 0, stream>>>(Ua1,   UUa1T, CH, CA, CH, CH);
        cvt_tr_kernel<<<(CH * CA + thr - 1) / thr, thr, 0, stream>>>(Wa1,   Wa1T,  CH, CA, 0, CH);
        init_dm_eos_kernel<<<(CL * CB + thr - 1) / thr, thr, 0, stream>>>(mask, dmIni, eosB, CL, CB);
    }

    // ---- embedding: Xe(L,B,H) = transpose(x @ W_emb + b_emb) ----
    {
        dim3 grid(CH / BN, (CB * CL) / BM);
        gemm_bf16_kernel<<<grid, 256, 0, stream>>>(
            x, CD, WembT, CD, CH, /*mode*/0, b_emb, XsA,
            /*xpose*/1, CL, CB, nullptr, nullptr, nullptr, 0, 0);
    }

    // ---- vertical layers ----
    for (int d = 0; d < CDEPTH; ++d) {
        const float* xsIn  = (d % 2 == 0) ? XsA : XsB;
        float*       hsOut = (d % 2 == 0) ? XsB : XsA;
        int* ActsOut = (d % 2 == 0) ? Acts0 : Acts1;
        int* NdmsOut = (d % 2 == 0) ? Ndms0 : Ndms1;
        const int* aprevActs = (d == 0) ? nullptr : ((d % 2 == 0) ? Acts1 : Acts0);
        const int* dmsPrev   = (d == 0) ? dmIni   : ((d % 2 == 0) ? Ndms1 : Ndms0);
        const int isLast = (d == CDEPTH - 1) ? 1 : 0;

        init_state_kernel<<<(CB * CH + 255) / 256, 256, 0, stream>>>(hCur, actCur, dmCur, CB * CH, CB);

        // hoisted input projections over all timesteps
        {
            dim3 gridW(CH / BN, (CL * CB) / BM);
            gemm_bf16_kernel<<<gridW, 256, 0, stream>>>(
                xsIn, CH, WT, CH, CH, 0, b, Xw, 0, CL, CB, nullptr, nullptr, nullptr, 0, 0);
            dim3 gridA(CA / BN, (CL * CB) / BM);
            gemm_bf16_kernel<<<gridA, 256, 0, stream>>>(
                xsIn, CH, Wa1T, CH, CA, 0, ba1, Xa, 0, CL, CB, nullptr, nullptr, nullptr, 0, 0);
        }

        // sequential horizontal scan
        for (int t = 0; t < CL; ++t) {
            dim3 gridS((CH + CA) / BN, CB / BM);   // 5 x 4
            gemm_bf16_kernel<<<gridS, 256, 0, stream>>>(
                hCur, CH, UUa1T, CH, CH + CA, /*mode*/1, nullptr, Hn,
                0, CL, CB,
                Xw + (size_t)t * CB * CH, Xa + (size_t)t * CB * CA, Pa, CH, CA);

            act_gate_kernel<<<CB / 8, 256, 0, stream>>>(
                Pa, Hn, xsIn, Wa2, ba2, aprevActs, dmsPrev, eosB,
                actCur, dmCur, ActsOut, NdmsOut, hCur, hsOut,
                t, CL, CB, CH, CA, isLast);
        }
    }

    // hs[-1] of last layer == final hCur
    copy_out_kernel<<<(CB * CH + 255) / 256, 256, 0, stream>>>(hCur, out, CB * CH);
}
